// TheoryModel_42777874268561
// MI455X (gfx1250) — matrix operand order repaired and checked by reading, not compiled, not run on hardware
//
#include <hip/hip_runtime.h>
#include <stdint.h>

typedef unsigned short u16;
typedef __attribute__((ext_vector_type(16))) __bf16          v16bf;
typedef __attribute__((ext_vector_type(8)))  float           v8f;
typedef __attribute__((ext_vector_type(4)))  unsigned int    u32x4;
typedef __attribute__((ext_vector_type(4)))  unsigned short  u16x4;

__device__ __forceinline__ u16 f2bf(float f) {
  unsigned int u = __float_as_uint(f);
  u += 0x7fffu + ((u >> 16) & 1u);      // round-to-nearest-even
  return (u16)(u >> 16);
}

// Low 32 bits of a flat shared-memory pointer == byte offset from the wave's
// LDS base (flat LDS aperture keeps the offset in addr[31:0]).
#define LDS_OFF(p) ((unsigned)(size_t)(p))

// CDNA5 async global->LDS copy, 16B per lane, tracked with ASYNCcnt.
// offset: immediate applies to BOTH the global and the LDS address.
__device__ __forceinline__ void async_ld16(unsigned lds, const void* g) {
  asm volatile("global_load_async_to_lds_b128 %0, %1, off"
               :: "v"(lds), "v"(g) : "memory");
}
__device__ __forceinline__ void async_ld16_o16(unsigned lds, const void* g) {
  asm volatile("global_load_async_to_lds_b128 %0, %1, off offset:16"
               :: "v"(lds), "v"(g) : "memory");
}
__device__ __forceinline__ void wait_async0() {
  asm volatile("s_wait_asynccnt 0x0" ::: "memory");
}

// ---------------------------------------------------------------- fp32 -> bf16
__global__ __launch_bounds__(256) void cvt_bf16_kernel(
    const float* __restrict__ in, u16* __restrict__ out, int n4) {
  int i = blockIdx.x * 256 + threadIdx.x;
  if (i < n4) {
    float4 v = ((const float4*)in)[i];
    u16x4 o;
    o.x = f2bf(v.x); o.y = f2bf(v.y); o.z = f2bf(v.z); o.w = f2bf(v.w);
    ((u16x4*)out)[i] = o;
  }
}

// ---------------------------------------------------------------- LayerNorm
__global__ __launch_bounds__(256) void ln_bf16_kernel(
    const float* __restrict__ x, const float* __restrict__ g,
    const float* __restrict__ b, u16* __restrict__ out) {
  const int row = blockIdx.x;
  const int tid = threadIdx.x;
  const int wave = tid >> 5, lane = tid & 31;
  const float4 v = ((const float4*)(x + (size_t)row * 1024))[tid];
  __shared__ float red[8];

  float s = v.x + v.y + v.z + v.w;
  #pragma unroll
  for (int o = 16; o > 0; o >>= 1) s += __shfl_xor(s, o, 32);
  if (lane == 0) red[wave] = s;
  __syncthreads();
  float tot = 0.f;
  #pragma unroll
  for (int i = 0; i < 8; ++i) tot += red[i];
  const float mean = tot * (1.0f / 1024.0f);

  const float dx = v.x - mean, dy = v.y - mean, dz = v.z - mean, dw = v.w - mean;
  float s2 = dx * dx + dy * dy + dz * dz + dw * dw;
  #pragma unroll
  for (int o = 16; o > 0; o >>= 1) s2 += __shfl_xor(s2, o, 32);
  __syncthreads();
  if (lane == 0) red[wave] = s2;
  __syncthreads();
  float tot2 = 0.f;
  #pragma unroll
  for (int i = 0; i < 8; ++i) tot2 += red[i];
  const float rstd = rsqrtf(tot2 * (1.0f / 1024.0f) + 1e-5f);

  const float4 gv = ((const float4*)g)[tid];
  const float4 bv = ((const float4*)b)[tid];
  u16x4 o;
  o.x = f2bf(dx * rstd * gv.x + bv.x);
  o.y = f2bf(dy * rstd * gv.y + bv.y);
  o.z = f2bf(dz * rstd * gv.z + bv.z);
  o.w = f2bf(dw * rstd * gv.w + bv.w);
  ((u16x4*)(out + (size_t)row * 1024))[tid] = o;
}

// ---------------------------------------------------------------- WMMA GEMM
// C[m,n] = alpha * sum_k A[m,k]*B[n,k]  (+bias[n]) (+causal -999) (ReLU) (+resid)
// A: M x K bf16 row-major, B: N x K bf16 row-major. M % 128, N % 64, K % 64 == 0.
// Block 256 thr / 8 waves; block tile 128x64x32; wave tile 64x16 (4 WMMA/step).
// Double-buffered LDS filled by async global->LDS DMA (ASYNCcnt), pipeline
// unrolled 2x (nk even) so each phase uses a fixed buffer.
union FragU { u32x4 q[2]; v16bf v; };

__device__ __forceinline__ void issue_panel(unsigned lA, unsigned lB,
    const u16* gA, const u16* gB, int k0) {
  async_ld16(lA, gA + k0);
  async_ld16_o16(lA, gA + k0);
  async_ld16(lB, gB + k0);
}

__device__ __forceinline__ void compute_panel(const u16* __restrict__ sA,
    const u16* __restrict__ sB, int wm, int wn, int r, int half, v8f (&acc)[4]) {
  // B fragment (ISA 7.12.2): lane = N col, K chunks {half*8..+7} and {16+half*8..+7}
  FragU fb, fa0, fa1, fa2, fa3;
  const int nr = wn + r;
  fb.q[0] = *(const u32x4*)&sB[nr * 40 + half * 8];
  fb.q[1] = *(const u32x4*)&sB[nr * 40 + 16 + half * 8];
  // A fragments: lane = M row, K chunks {half*8..+7} and {16+half*8..+7}
  const int ar = wm + r;
  fa0.q[0] = *(const u32x4*)&sA[(ar +  0) * 40 + half * 8];
  fa0.q[1] = *(const u32x4*)&sA[(ar +  0) * 40 + 16 + half * 8];
  fa1.q[0] = *(const u32x4*)&sA[(ar + 16) * 40 + half * 8];
  fa1.q[1] = *(const u32x4*)&sA[(ar + 16) * 40 + 16 + half * 8];
  fa2.q[0] = *(const u32x4*)&sA[(ar + 32) * 40 + half * 8];
  fa2.q[1] = *(const u32x4*)&sA[(ar + 32) * 40 + 16 + half * 8];
  fa3.q[0] = *(const u32x4*)&sA[(ar + 48) * 40 + half * 8];
  fa3.q[1] = *(const u32x4*)&sA[(ar + 48) * 40 + 16 + half * 8];
  acc[0] = __builtin_amdgcn_wmma_f32_16x16x32_bf16(
      false, fa0.v, false, fb.v, (short)0, acc[0], false, false);
  acc[1] = __builtin_amdgcn_wmma_f32_16x16x32_bf16(
      false, fa1.v, false, fb.v, (short)0, acc[1], false, false);
  acc[2] = __builtin_amdgcn_wmma_f32_16x16x32_bf16(
      false, fa2.v, false, fb.v, (short)0, acc[2], false, false);
  acc[3] = __builtin_amdgcn_wmma_f32_16x16x32_bf16(
      false, fa3.v, false, fb.v, (short)0, acc[3], false, false);
}

template<bool OUTBF16, bool RELU, bool CAUSAL>
__global__ __launch_bounds__(256) void gemm_nt_kernel(
    const u16* __restrict__ Ag, const u16* __restrict__ Bg,
    const float* __restrict__ bias, const float* __restrict__ resid,
    void* __restrict__ Cg, int M, int N, int K, float alpha,
    long long strA, long long strB, long long strC, long long strR) {
  __shared__ u16 shA[2][128 * 40];   // 128 rows x 32 K, pad to 40 (16B rows)
  __shared__ u16 shB[2][64 * 40];

  const int tid  = threadIdx.x;
  const int wave = tid >> 5, lane = tid & 31;
  const int r    = lane & 15, half = lane >> 4;
  const int bm = blockIdx.x * 128, bn = blockIdx.y * 64, bz = blockIdx.z;
  const u16* A = Ag + (size_t)bz * strA;
  const u16* B = Bg + (size_t)bz * strB;

  const int wm = (wave >> 2) * 64;  // wave rows: 0 or 64 (4 x 16 tiles each)
  const int wn = (wave & 3) * 16;   // wave cols: 0,16,32,48

  // async staging: A 128x32 -> 2 x 16B per thread; B 64x32 -> 1 x 16B
  const int arow = tid >> 1, acol = (tid & 1) * 16;   // acol in elements
  const int brow = tid >> 2, bcol = (tid & 3) * 8;
  const u16* gA = A + (size_t)(bm + arow) * K + acol;
  const u16* gB = B + (size_t)(bn + brow) * K + bcol;
  const unsigned lA0 = LDS_OFF(&shA[0][arow * 40 + acol]);
  const unsigned lA1 = LDS_OFF(&shA[1][arow * 40 + acol]);
  const unsigned lB0 = LDS_OFF(&shB[0][brow * 40 + bcol]);
  const unsigned lB1 = LDS_OFF(&shB[1][brow * 40 + bcol]);

  v8f acc[4] = {};

  const int nk = K >> 5;            // even for all shapes used here
  issue_panel(lA0, lB0, gA, gB, 0); // prologue: panel 0 -> buf0
  wait_async0();
  __syncthreads();

  for (int it = 0; it < nk; it += 2) {
    // phase 0: prefetch panel it+1 -> buf1, compute buf0
    issue_panel(lA1, lB1, gA, gB, (it + 1) << 5);
    compute_panel(shA[0], shB[0], wm, wn, r, half, acc);
    wait_async0();
    __syncthreads();
    // phase 1: prefetch panel it+2 -> buf0, compute buf1
    if (it + 2 < nk) issue_panel(lA0, lB0, gA, gB, (it + 2) << 5);
    compute_panel(shA[1], shB[1], wm, wn, r, half, acc);
    wait_async0();
    __syncthreads();
  }

  // Epilogue: C/D layout — lane = N col; VGPR j = row half*8 + j of 16-tile.
  const int ng = bn + wn + r;
  const float bia = bias ? bias[ng] : 0.0f;
  const float* Rp = resid ? (resid + (size_t)bz * strR) : nullptr;
  float* Cf = (float*)Cg + (size_t)bz * strC;
  u16*   Cb = (u16*)Cg + (size_t)bz * strC;

  #pragma unroll
  for (int t = 0; t < 4; ++t) {
    const int rowbase = bm + wm + t * 16 + half * 8;
    #pragma unroll
    for (int j = 0; j < 8; ++j) {
      const int mg = rowbase + j;
      float vv = acc[t][j] * alpha + bia;
      if (CAUSAL && ng > mg) vv += -999.0f;
      if (RELU) vv = fmaxf(vv, 0.0f);
      if (Rp) vv += Rp[(size_t)mg * N + ng];
      if (OUTBF16) Cb[(size_t)mg * N + ng] = f2bf(vv);
      else         Cf[(size_t)mg * N + ng] = vv;
    }
  }
}

// ---------------------------------------------------------------- softmax
__global__ __launch_bounds__(256) void softmax_kernel(
    float* __restrict__ S, u16* __restrict__ P) {
  const int row = blockIdx.x;
  float* sr = S + (size_t)row * 2048;
  u16*   pr = P + (size_t)row * 2048;
  const int tid = threadIdx.x;
  const int wave = tid >> 5, lane = tid & 31;
  __shared__ float red[8];

  float4 a = ((float4*)sr)[tid];
  float4 b = ((float4*)sr)[tid + 256];
  float mx = fmaxf(fmaxf(fmaxf(a.x, a.y), fmaxf(a.z, a.w)),
                   fmaxf(fmaxf(b.x, b.y), fmaxf(b.z, b.w)));
  #pragma unroll
  for (int o = 16; o > 0; o >>= 1) mx = fmaxf(mx, __shfl_xor(mx, o, 32));
  if (lane == 0) red[wave] = mx;
  __syncthreads();
  float m = red[0];
  #pragma unroll
  for (int i = 1; i < 8; ++i) m = fmaxf(m, red[i]);

  a.x = __expf(a.x - m); a.y = __expf(a.y - m);
  a.z = __expf(a.z - m); a.w = __expf(a.w - m);
  b.x = __expf(b.x - m); b.y = __expf(b.y - m);
  b.z = __expf(b.z - m); b.w = __expf(b.w - m);
  float s = a.x + a.y + a.z + a.w + b.x + b.y + b.z + b.w;
  #pragma unroll
  for (int o = 16; o > 0; o >>= 1) s += __shfl_xor(s, o, 32);
  __syncthreads();
  if (lane == 0) red[wave] = s;
  __syncthreads();
  float tot = 0.f;
  #pragma unroll
  for (int i = 0; i < 8; ++i) tot += red[i];
  const float inv = 1.0f / tot;

  a.x *= inv; a.y *= inv; a.z *= inv; a.w *= inv;
  b.x *= inv; b.y *= inv; b.z *= inv; b.w *= inv;
  ((float4*)sr)[tid]       = a;
  ((float4*)sr)[tid + 256] = b;
  u16x4 oa, ob;
  oa.x = f2bf(a.x); oa.y = f2bf(a.y); oa.z = f2bf(a.z); oa.w = f2bf(a.w);
  ob.x = f2bf(b.x); ob.y = f2bf(b.y); ob.z = f2bf(b.z); ob.w = f2bf(b.w);
  ((u16x4*)pr)[tid]       = oa;
  ((u16x4*)pr)[tid + 256] = ob;
}

// ---------------------------------------------------------------- V transpose
__global__ __launch_bounds__(256) void transpose_v_kernel(
    const u16* __restrict__ V, u16* __restrict__ VT) {
  const int idx = blockIdx.x * 256 + threadIdx.x;   // over B*D*L = 8388608
  const int l = idx & 2047;
  const int e = (idx >> 11) & 1023;
  const int b = idx >> 21;
  VT[idx] = V[((size_t)b << 21) + ((size_t)l << 10) + e];
}

// ---------------------------------------------------------------- launch
extern "C" void kernel_launch(void* const* d_in, const int* in_sizes, int n_in,
                              void* d_out, int out_size, void* d_ws, size_t ws_size,
                              hipStream_t stream) {
  (void)in_sizes; (void)n_in; (void)out_size; (void)ws_size;
  const float* x   = (const float*)d_in[0];
  const float* Wq  = (const float*)d_in[1];
  const float* bq  = (const float*)d_in[2];
  const float* Wk  = (const float*)d_in[3];
  const float* bk  = (const float*)d_in[4];
  const float* Wv  = (const float*)d_in[5];
  const float* bv  = (const float*)d_in[6];
  const float* W1  = (const float*)d_in[7];
  const float* b1  = (const float*)d_in[8];
  const float* W2  = (const float*)d_in[9];
  const float* b2  = (const float*)d_in[10];
  const float* g1  = (const float*)d_in[11];
  const float* be1 = (const float*)d_in[12];
  const float* g2  = (const float*)d_in[13];
  const float* be2 = (const float*)d_in[14];

  constexpr int  Bb = 4, L = 2048, D = 1024, F = 4096;
  constexpr size_t BL = (size_t)Bb * L;         // 8192 rows
  float* logits = (float*)d_out;                // BL*D fp32
  float* wts    = logits + BL * D;              // B*L*L fp32 (scores -> softmax in place)

  char* w = (char*)d_ws;
  auto alloc = [&](size_t bytes) { char* p = w; w += (bytes + 255) & ~(size_t)255; return p; };
  u16*   h_bf  = (u16*)alloc(BL * D * 2);       // LN1 out; reused for LN2 out
  u16*   Q_bf  = (u16*)alloc(BL * D * 2);
  u16*   K_bf  = (u16*)alloc(BL * D * 2);
  u16*   V_bf  = (u16*)alloc(BL * D * 2);
  u16*   VT_bf = (u16*)alloc(BL * D * 2);
  u16*   P_bf  = (u16*)alloc((size_t)Bb * L * L * 2);
  float* x2    = (float*)alloc(BL * D * 4);     // x + attn_out
  u16*   F_bf  = (u16*)alloc(BL * F * 2);       // relu(ffn1)
  u16*   Wq_bf = (u16*)alloc((size_t)D * D * 2);
  u16*   Wk_bf = (u16*)alloc((size_t)D * D * 2);
  u16*   Wv_bf = (u16*)alloc((size_t)D * D * 2);
  u16*   W1_bf = (u16*)alloc((size_t)F * D * 2);
  u16*   W2_bf = (u16*)alloc((size_t)D * F * 2);

  const dim3 TB(256);
  cvt_bf16_kernel<<<dim3(D * D / 4 / 256), TB, 0, stream>>>(Wq, Wq_bf, D * D / 4);
  cvt_bf16_kernel<<<dim3(D * D / 4 / 256), TB, 0, stream>>>(Wk, Wk_bf, D * D / 4);
  cvt_bf16_kernel<<<dim3(D * D / 4 / 256), TB, 0, stream>>>(Wv, Wv_bf, D * D / 4);
  cvt_bf16_kernel<<<dim3(F * D / 4 / 256), TB, 0, stream>>>(W1, W1_bf, F * D / 4);
  cvt_bf16_kernel<<<dim3(D * F / 4 / 256), TB, 0, stream>>>(W2, W2_bf, D * F / 4);

  // LN1: h = LN(x)
  ln_bf16_kernel<<<dim3((unsigned)BL), TB, 0, stream>>>(x, g1, be1, h_bf);

  // Q/K/V = h*W^T + b (bf16 out)
  gemm_nt_kernel<true, false, false><<<dim3(64, 16, 1), TB, 0, stream>>>(
      h_bf, Wq_bf, bq, nullptr, Q_bf, 8192, 1024, 1024, 1.0f, 0, 0, 0, 0);
  gemm_nt_kernel<true, false, false><<<dim3(64, 16, 1), TB, 0, stream>>>(
      h_bf, Wk_bf, bk, nullptr, K_bf, 8192, 1024, 1024, 1.0f, 0, 0, 0, 0);
  gemm_nt_kernel<true, false, false><<<dim3(64, 16, 1), TB, 0, stream>>>(
      h_bf, Wv_bf, bv, nullptr, V_bf, 8192, 1024, 1024, 1.0f, 0, 0, 0, 0);

  // scores = Q*K^T / 32 + causal(-999) -> d_out wts region
  gemm_nt_kernel<false, false, true><<<dim3(16, 32, 4), TB, 0, stream>>>(
      Q_bf, K_bf, nullptr, nullptr, wts, 2048, 2048, 1024, 0.03125f,
      (long long)L * D, (long long)L * D, (long long)L * L, 0);

  // softmax in place (fp32) + bf16 copy for PV
  softmax_kernel<<<dim3((unsigned)BL), TB, 0, stream>>>(wts, P_bf);

  // V^T for NT GEMM
  transpose_v_kernel<<<dim3(32768), TB, 0, stream>>>(V_bf, VT_bf);

  // x2 = x + P*V
  gemm_nt_kernel<false, false, false><<<dim3(16, 16, 4), TB, 0, stream>>>(
      P_bf, VT_bf, nullptr, x, x2, 2048, 1024, 2048, 1.0f,
      (long long)L * L, (long long)D * L, (long long)L * D, (long long)L * D);

  // LN2: h2 = LN(x2) (reuse h_bf)
  ln_bf16_kernel<<<dim3((unsigned)BL), TB, 0, stream>>>(x2, g2, be2, h_bf);

  // FFN1: F = relu(h2*W1^T + b1) (bf16 out)
  gemm_nt_kernel<true, true, false><<<dim3(64, 64, 1), TB, 0, stream>>>(
      h_bf, W1_bf, b1, nullptr, F_bf, 8192, 4096, 1024, 1.0f, 0, 0, 0, 0);

  // FFN2: logits = x2 + F*W2^T + b2
  gemm_nt_kernel<false, false, false><<<dim3(64, 16, 1), TB, 0, stream>>>(
      F_bf, W2_bf, b2, x2, logits, 8192, 1024, 4096, 1.0f, 0, 0, 0, 0);
}
